// ChildSumTreeLSTM_12180527251607
// MI455X (gfx1250) — compile-verified, matrix operand order checked by reading, not run on hardware
//
#include <hip/hip_runtime.h>
#include <hip/hip_bf16.h>

// ---------------- problem constants ----------------
#define BB    32
#define DD    11
#define NN    2047           // nodes per tree
#define MM    300            // IN_DIM == MEM_DIM
#define KPAD  320            // K padded to 10 * 32
#define NPAD  304            // N padded to 19 * 16
#define LDG   304            // row stride (floats) of gate / c buffers
#define ROWS  (BB * NN)      // 65504 (multiple of 32)
#define NTILES 19
#define WPB   4              // waves per block in GEMM kernels
#define KSTEPS (KPAD / 32)   // 10

typedef __attribute__((ext_vector_type(8)))  __bf16 v8bf;
typedef __attribute__((ext_vector_type(16))) __bf16 v16bf;
typedef __attribute__((ext_vector_type(8)))  float  v8f;

__device__ __forceinline__ __bf16 f2bf(float f) {
    unsigned u = __builtin_bit_cast(unsigned, f);
    unsigned r = u + 0x7FFFu + ((u >> 16) & 1u);   // round-to-nearest-even
    return __builtin_bit_cast(__bf16, (unsigned short)(r >> 16));
}
__device__ __forceinline__ float sigm(float x) { return 1.0f / (1.0f + __expf(-x)); }

// A/B fragment per ISA 7.12.2: lane holds row (A) / col (B) = lane&15;
// elems 0..7 = K(khalf*8 .. +7), elems 8..15 = K(16+khalf*8 .. +7), khalf = lane>>4.
__device__ __forceinline__ v16bf load_frag(const __bf16* p) {
    v8bf a = *(const v8bf*)(p);
    v8bf b = *(const v8bf*)(p + 16);
    return __builtin_shufflevector(a, b, 0,1,2,3,4,5,6,7,8,9,10,11,12,13,14,15);
}

#define WMMA_BF16(A, Bm, C) \
    __builtin_amdgcn_wmma_f32_16x16x32_bf16(false, (A), false, (Bm), (short)0, (C), false, false)

// ---------------- prep kernels ----------------
__global__ void prep_x_kernel(const float* __restrict__ embs, __bf16* __restrict__ X) {
    const size_t total = (size_t)ROWS * KPAD;
    for (size_t idx = (size_t)blockIdx.x * blockDim.x + threadIdx.x; idx < total;
         idx += (size_t)gridDim.x * blockDim.x) {
        int k = (int)(idx % KPAD);
        size_t r = idx / KPAD;
        X[idx] = (k < MM) ? f2bf(embs[r * MM + k])
                          : __builtin_bit_cast(__bf16, (unsigned short)0);
    }
}

// Wt slot order: 0=Wix 1=Wfx 2=Wox 3=Wux 4=Wih 5=Wfh 6=Woh 7=Wuh
// Wt[s][n*KPAD + k] = W[k*MM + n]  (transposed, bf16, zero padded)
__global__ void prep_w_kernel(const float* __restrict__ W0, const float* __restrict__ W1,
                              const float* __restrict__ W2, const float* __restrict__ W3,
                              const float* __restrict__ W4, const float* __restrict__ W5,
                              const float* __restrict__ W6, const float* __restrict__ W7,
                              __bf16* __restrict__ Wt) {
    const size_t per = (size_t)NPAD * KPAD, total = 8 * per;
    for (size_t idx = (size_t)blockIdx.x * blockDim.x + threadIdx.x; idx < total;
         idx += (size_t)gridDim.x * blockDim.x) {
        int s = (int)(idx / per);
        size_t rem = idx % per;
        int n = (int)(rem / KPAD), k = (int)(rem % KPAD);
        const float* W = (s == 0) ? W0 : (s == 1) ? W1 : (s == 2) ? W2 : (s == 3) ? W3
                       : (s == 4) ? W4 : (s == 5) ? W5 : (s == 6) ? W6 : W7;
        float v = (n < MM && k < MM) ? W[(size_t)k * MM + n] : 0.0f;
        Wt[idx] = f2bf(v);
    }
}

// bcomb[g][n] = b{g}x[n] + b{g}h[n], g: 0=i 1=f 2=o 3=u
__global__ void prep_b_kernel(const float* __restrict__ bix, const float* __restrict__ bih,
                              const float* __restrict__ bfx, const float* __restrict__ bfh,
                              const float* __restrict__ box, const float* __restrict__ boh,
                              const float* __restrict__ bux, const float* __restrict__ buh,
                              float* __restrict__ bcomb) {
    int idx = blockIdx.x * blockDim.x + threadIdx.x;
    if (idx >= 4 * NPAD) return;
    int g = idx / NPAD, n = idx % NPAD;
    float v = 0.0f;
    if (n < MM) {
        const float* bx = (g == 0) ? bix : (g == 1) ? bfx : (g == 2) ? box : bux;
        const float* bh = (g == 0) ? bih : (g == 1) ? bfh : (g == 2) ? boh : buh;
        v = bx[n] + bh[n];
    }
    bcomb[idx] = v;
}

// ---------------- fused 4-gate x-projection GEMM (2 M-tiles / wave) ----------------
__global__ void __launch_bounds__(WPB * 32)
gemm_x_kernel(const __bf16* __restrict__ X, const __bf16* __restrict__ Wt,
              const float* __restrict__ bcomb,
              float* __restrict__ gi, float* __restrict__ gf,
              float* __restrict__ go, float* __restrict__ gu) {
    const int wave = threadIdx.x >> 5, lane = threadIdx.x & 31;
    const int pair = blockIdx.x * WPB + wave;
    const int PAIRS = (ROWS / 16) / 2;       // 2047
    if (pair >= PAIRS) return;               // wave-uniform
    const int mt0 = pair * 2;
    const int ntile = blockIdx.y;
    const int ar = lane & 15, kh = lane >> 4;

    const __bf16* ap0 = X + (size_t)(mt0 * 16 + ar) * KPAD + kh * 8;
    const __bf16* ap1 = ap0 + (size_t)16 * KPAD;
    const size_t wrow = (size_t)(ntile * 16 + ar) * KPAD + kh * 8;
    const __bf16* bi = Wt + (size_t)0 * NPAD * KPAD + wrow;
    const __bf16* bf = Wt + (size_t)1 * NPAD * KPAD + wrow;
    const __bf16* bo = Wt + (size_t)2 * NPAD * KPAD + wrow;
    const __bf16* bu = Wt + (size_t)3 * NPAD * KPAD + wrow;

    v8f ci0 = {}, cf0 = {}, co0 = {}, cu0 = {};
    v8f ci1 = {}, cf1 = {}, co1 = {}, cu1 = {};
    for (int kt = 0; kt < KSTEPS; ++kt) {
        v16bf A0 = load_frag(ap0 + kt * 32);
        v16bf A1 = load_frag(ap1 + kt * 32);
        if (kt + 1 < KSTEPS) {               // gfx1250 global_prefetch_b8
            __builtin_prefetch((const void*)(ap0 + (kt + 1) * 32), 0, 3);
            __builtin_prefetch((const void*)(ap1 + (kt + 1) * 32), 0, 3);
        }
        v16bf Bi = load_frag(bi + kt * 32);
        ci0 = WMMA_BF16(A0, Bi, ci0);
        ci1 = WMMA_BF16(A1, Bi, ci1);
        v16bf Bf = load_frag(bf + kt * 32);
        cf0 = WMMA_BF16(A0, Bf, cf0);
        cf1 = WMMA_BF16(A1, Bf, cf1);
        v16bf Bo = load_frag(bo + kt * 32);
        co0 = WMMA_BF16(A0, Bo, co0);
        co1 = WMMA_BF16(A1, Bo, co1);
        v16bf Bu = load_frag(bu + kt * 32);
        cu0 = WMMA_BF16(A0, Bu, cu0);
        cu1 = WMMA_BF16(A1, Bu, cu1);
    }
    const int n = ntile * 16 + ar;           // always < NPAD
    const int mb = kh * 8;
    const float bci = bcomb[0 * NPAD + n], bcf = bcomb[1 * NPAD + n];
    const float bco = bcomb[2 * NPAD + n], bcu = bcomb[3 * NPAD + n];
#pragma unroll
    for (int t = 0; t < 2; ++t) {
#pragma unroll
        for (int v = 0; v < 8; ++v) {
            size_t r = (size_t)((mt0 + t) * 16 + mb + v);
            gi[r * LDG + n] = (t ? ci1[v] : ci0[v]) + bci;
            gf[r * LDG + n] = (t ? cf1[v] : cf0[v]) + bcf;
            go[r * LDG + n] = (t ? co1[v] : co0[v]) + bco;
            gu[r * LDG + n] = (t ? cu1[v] : cu0[v]) + bcu;
        }
    }
}

// ---------------- leaf level (elementwise) ----------------
__global__ void leaf_kernel(const float* __restrict__ gi, const float* __restrict__ go,
                            const float* __restrict__ gu,
                            float* __restrict__ cbuf, float* __restrict__ hout) {
    const int NL = 1 << (DD - 1);            // 1024 leaves
    const size_t total = (size_t)BB * NL * MM;
    for (size_t idx = (size_t)blockIdx.x * blockDim.x + threadIdx.x; idx < total;
         idx += (size_t)gridDim.x * blockDim.x) {
        int n = (int)(idx % MM);
        size_t t = idx / MM;
        int b = (int)(t / NL), ii = (int)(t % NL);
        size_t rf = (size_t)b * NN + (NL - 1 + ii);
        float iv = sigm(gi[rf * LDG + n]);
        float ov = sigm(go[rf * LDG + n]);
        float uv = tanhf(gu[rf * LDG + n]);
        float cl = iv * uv;
        cbuf[rf * LDG + n] = cl;
        hout[rf * MM + n] = ov * tanhf(cl);
    }
}

// ---------------- per-level child gather / pack to bf16 ----------------
__global__ void pack_kernel(const float* __restrict__ h, __bf16* __restrict__ Hs,
                            __bf16* __restrict__ H1, __bf16* __restrict__ H2,
                            int l, int nl, int R) {
    const size_t total = (size_t)R * KPAD;
    for (size_t idx = (size_t)blockIdx.x * blockDim.x + threadIdx.x; idx < total;
         idx += (size_t)gridDim.x * blockDim.x) {
        int k = (int)(idx % KPAD);
        int r = (int)(idx / KPAD);
        int b = r >> l, ii = r & (nl - 1);
        int j = (nl - 1) + ii;
        float v1 = 0.0f, v2 = 0.0f;
        if (k < MM) {
            v1 = h[((size_t)b * NN + (2 * j + 1)) * MM + k];
            v2 = h[((size_t)b * NN + (2 * j + 2)) * MM + k];
        }
        Hs[idx] = f2bf(v1 + v2);
        H1[idx] = f2bf(v1);
        H2[idx] = f2bf(v2);
    }
}

// ---------------- per-level recurrent GEMM + fused gates (2 M-tiles / wave) ----------------
__global__ void __launch_bounds__(WPB * 32)
gemm_r_kernel(const __bf16* __restrict__ Hs, const __bf16* __restrict__ H1,
              const __bf16* __restrict__ H2, const __bf16* __restrict__ Wt,
              const float* __restrict__ gi, const float* __restrict__ gf,
              const float* __restrict__ go, const float* __restrict__ gu,
              float* __restrict__ cbuf, float* __restrict__ hout,
              int l, int nl, int R) {
    const int wave = threadIdx.x >> 5, lane = threadIdx.x & 31;
    const int pair = blockIdx.x * WPB + wave;
    const int PAIRS = (R / 16) / 2;          // R = 32*2^l -> always even tile count
    if (pair >= PAIRS) return;               // wave-uniform
    const int mt0 = pair * 2;
    const int ntile = blockIdx.y;
    const int ar = lane & 15, kh = lane >> 4;

    const size_t arow = (size_t)(mt0 * 16 + ar) * KPAD + kh * 8;
    const size_t astep = (size_t)16 * KPAD;
    const __bf16* as = Hs + arow;
    const __bf16* a1 = H1 + arow;
    const __bf16* a2 = H2 + arow;
    const size_t wrow = (size_t)(ntile * 16 + ar) * KPAD + kh * 8;
    const __bf16* wi = Wt + (size_t)4 * NPAD * KPAD + wrow;   // Wih
    const __bf16* wf = Wt + (size_t)5 * NPAD * KPAD + wrow;   // Wfh
    const __bf16* wo = Wt + (size_t)6 * NPAD * KPAD + wrow;   // Woh
    const __bf16* wu = Wt + (size_t)7 * NPAD * KPAD + wrow;   // Wuh

    v8f Ai0 = {}, Ao0 = {}, Au0 = {}, F10 = {}, F20 = {};
    v8f Ai1 = {}, Ao1 = {}, Au1 = {}, F11 = {}, F21 = {};
    for (int kt = 0; kt < KSTEPS; ++kt) {
        v16bf As0 = load_frag(as + kt * 32);
        v16bf As1 = load_frag(as + astep + kt * 32);
        v16bf Wi = load_frag(wi + kt * 32);
        Ai0 = WMMA_BF16(As0, Wi, Ai0);
        Ai1 = WMMA_BF16(As1, Wi, Ai1);
        v16bf Wo = load_frag(wo + kt * 32);
        Ao0 = WMMA_BF16(As0, Wo, Ao0);
        Ao1 = WMMA_BF16(As1, Wo, Ao1);
        v16bf Wu = load_frag(wu + kt * 32);
        Au0 = WMMA_BF16(As0, Wu, Au0);
        Au1 = WMMA_BF16(As1, Wu, Au1);
        v16bf Wf = load_frag(wf + kt * 32);
        v16bf A10 = load_frag(a1 + kt * 32);
        v16bf A11 = load_frag(a1 + astep + kt * 32);
        F10 = WMMA_BF16(A10, Wf, F10);
        F11 = WMMA_BF16(A11, Wf, F11);
        v16bf A20 = load_frag(a2 + kt * 32);
        v16bf A21 = load_frag(a2 + astep + kt * 32);
        F20 = WMMA_BF16(A20, Wf, F20);
        F21 = WMMA_BF16(A21, Wf, F21);
        if (kt + 1 < KSTEPS) {               // gfx1250 global_prefetch_b8
            __builtin_prefetch((const void*)(as + (kt + 1) * 32), 0, 3);
            __builtin_prefetch((const void*)(a1 + (kt + 1) * 32), 0, 3);
            __builtin_prefetch((const void*)(a2 + (kt + 1) * 32), 0, 3);
        }
    }
    const int n = ntile * 16 + ar;
    if (n < MM) {                            // divergence only after WMMAs
#pragma unroll
        for (int t = 0; t < 2; ++t) {
#pragma unroll
            for (int v = 0; v < 8; ++v) {
                int r = (mt0 + t) * 16 + kh * 8 + v;
                int b = r >> l, ii = r & (nl - 1);
                int j = (nl - 1) + ii;
                size_t rf = (size_t)b * NN + j;
                float iv = sigm(gi[rf * LDG + n] + (t ? Ai1[v] : Ai0[v]));
                float ov = sigm(go[rf * LDG + n] + (t ? Ao1[v] : Ao0[v]));
                float uv = tanhf(gu[rf * LDG + n] + (t ? Au1[v] : Au0[v]));
                float fx = gf[rf * LDG + n];
                float f1 = sigm(fx + (t ? F11[v] : F10[v]));
                float f2 = sigm(fx + (t ? F21[v] : F20[v]));
                float c1 = cbuf[((size_t)b * NN + (2 * j + 1)) * LDG + n];
                float c2 = cbuf[((size_t)b * NN + (2 * j + 2)) * LDG + n];
                float cn = iv * uv + f1 * c1 + f2 * c2;
                cbuf[rf * LDG + n] = cn;
                hout[rf * MM + n] = ov * tanhf(cn);
            }
        }
    }
}

// ---------------- host launcher ----------------
extern "C" void kernel_launch(void* const* d_in, const int* in_sizes, int n_in,
                              void* d_out, int out_size, void* d_ws, size_t ws_size,
                              hipStream_t stream) {
    const float* embs = (const float*)d_in[0];
    const float* Wix = (const float*)d_in[1];  const float* bix = (const float*)d_in[2];
    const float* Wih = (const float*)d_in[3];  const float* bih = (const float*)d_in[4];
    const float* Wfx = (const float*)d_in[5];  const float* bfx = (const float*)d_in[6];
    const float* Wfh = (const float*)d_in[7];  const float* bfh = (const float*)d_in[8];
    const float* Wox = (const float*)d_in[9];  const float* box = (const float*)d_in[10];
    const float* Woh = (const float*)d_in[11]; const float* boh = (const float*)d_in[12];
    const float* Wux = (const float*)d_in[13]; const float* bux = (const float*)d_in[14];
    const float* Wuh = (const float*)d_in[15]; const float* buh = (const float*)d_in[16];
    float* hout = (float*)d_out;

    char* ws = (char*)d_ws;
    size_t off = 0;
    auto carve = [&](size_t bytes) { char* p = ws + off; off += (bytes + 255) & ~(size_t)255; return p; };
    __bf16* Xpad  = (__bf16*)carve((size_t)ROWS * KPAD * 2);
    __bf16* Wt    = (__bf16*)carve((size_t)8 * NPAD * KPAD * 2);
    float*  bcomb = (float*) carve((size_t)4 * NPAD * 4);
    float*  gi    = (float*) carve((size_t)ROWS * LDG * 4);
    float*  gf    = (float*) carve((size_t)ROWS * LDG * 4);
    float*  go    = (float*) carve((size_t)ROWS * LDG * 4);
    float*  gu    = (float*) carve((size_t)ROWS * LDG * 4);
    float*  cbuf  = (float*) carve((size_t)ROWS * LDG * 4);
    const int RMAX = BB * (1 << (DD - 2)) * 2;   // 32768 rows (leaf-parent pack uses R = 16384)
    __bf16* Hs = (__bf16*)carve((size_t)RMAX * KPAD * 2);
    __bf16* H1 = (__bf16*)carve((size_t)RMAX * KPAD * 2);
    __bf16* H2 = (__bf16*)carve((size_t)RMAX * KPAD * 2);
    (void)ws_size; (void)in_sizes; (void)n_in; (void)out_size;

    // prep
    prep_x_kernel<<<4096, 256, 0, stream>>>(embs, Xpad);
    prep_w_kernel<<<3072, 256, 0, stream>>>(Wix, Wfx, Wox, Wux, Wih, Wfh, Woh, Wuh, Wt);
    prep_b_kernel<<<(4 * NPAD + 255) / 256, 256, 0, stream>>>(bix, bih, bfx, bfh,
                                                              box, boh, bux, buh, bcomb);
    // fused 4-gate x projection
    {
        int pairs = (ROWS / 16) / 2;                 // 2047
        dim3 grid((pairs + WPB - 1) / WPB, NTILES);
        gemm_x_kernel<<<grid, WPB * 32, 0, stream>>>(Xpad, Wt, bcomb, gi, gf, go, gu);
    }
    // leaves
    leaf_kernel<<<4096, 256, 0, stream>>>(gi, go, gu, cbuf, hout);
    // serial tree levels
    for (int l = DD - 2; l >= 0; --l) {
        int nl = 1 << l;
        int R = BB * nl;
        size_t total = (size_t)R * KPAD;
        int pblocks = (int)((total + 255) / 256);
        if (pblocks > 4096) pblocks = 4096;
        pack_kernel<<<pblocks, 256, 0, stream>>>(hout, Hs, H1, H2, l, nl, R);
        int pairs = (R / 16) / 2;
        dim3 grid((pairs + WPB - 1) / WPB, NTILES);
        gemm_r_kernel<<<grid, WPB * 32, 0, stream>>>(Hs, H1, H2, Wt, gi, gf, go, gu,
                                                     cbuf, hout, l, nl, R);
    }
}
